// BHS_TEST_16724602651186
// MI455X (gfx1250) — matrix operand order repaired and checked by reading, not compiled, not run on hardware
//
#include <hip/hip_runtime.h>
#include <hip/hip_bf16.h>

typedef __attribute__((ext_vector_type(16))) __bf16 v16bf;
typedef __attribute__((ext_vector_type(8)))  float  v8f;

#define NN      8192
#define FINF    16
#define HD      64
#define EE      131072
#define SS      4
#define SNN     (SS*NN)          // 32768
#define FEATF   (NN*HD)          // 524288
#define NSLICES 33               // 32 data slices (f x jhalf) + 1 bias slice
#define NTILES  4                // 64 h columns / 16
#define BELEM   (NSLICES*NTILES*512)   // bf16 elements in packed B (67584)

// ---------------- zero init ----------------
__global__ void zero_kernel(float* __restrict__ p, int n) {
    int i = blockIdx.x * 256 + threadIdx.x;
    for (; i < n; i += gridDim.x * 256) p[i] = 0.0f;
}

// ---------------- pack B = permuted W2 (+bias slice) into WMMA B-layout ----------------
// element index: [tile][lane][t], tile = slice*4+ntile
// B 32x16 bf16 layout: lane L -> N = L&15, K = (t<8?t:t+8) + (L<16?0:8)
__global__ void prep_b_kernel(const float* __restrict__ W2, const float* __restrict__ b2,
                              __bf16* __restrict__ Bbuf) {
    int idx = blockIdx.x * 256 + threadIdx.x;
    if (idx >= BELEM) return;
    int t     = idx & 15;
    int lane  = (idx >> 4) & 31;
    int tile  = idx >> 9;
    int ntile = tile & 3;
    int slice = tile >> 2;
    int K = ((t < 8) ? t : t + 8) + ((lane < 16) ? 0 : 8);
    int h = ntile * 16 + (lane & 15);
    float v;
    if (slice < 32) {
        int f = slice >> 1;
        int j = ((slice & 1) << 5) + K;
        v = W2[(f * 64 + h) * 64 + j];      // W2 is (1024, 64) row-major, row = f*64+h
    } else {
        v = (K < 16) ? b2[K * 64 + h] : 0.0f;   // bias slice: A carries raw x[e,f]
    }
    Bbuf[idx] = (__bf16)v;
}

// ---------------- NNConv fused GEMM + scatter-add (WMMA bf16) ----------------
__global__ __launch_bounds__(256) void nnconv_kernel(
    const float* __restrict__ x, const int* __restrict__ edge,
    const float* __restrict__ ea, const float* __restrict__ W1,
    const float* __restrict__ b1, const __bf16* __restrict__ Bbuf,
    float* __restrict__ agg)
{
    __shared__ __align__(32) __bf16 Bl[BELEM];   // 132 KB (fits 320 KB WGP LDS)
    int tid = threadIdx.x;
    {
        const uint4* s = (const uint4*)Bbuf;
        uint4* d = (uint4*)Bl;
        #pragma unroll 4
        for (int i = tid; i < BELEM / 8; i += 256) d[i] = s[i];
    }
    __syncthreads();

    int wave = tid >> 5, lane = tid & 31;
    int hi   = (lane >> 4) & 1;
    int base = (blockIdx.x * 8 + wave) * 16;     // 16 edges per wave
    int e    = base + (lane & 15);

    // eh = relu(ea*W1 + b1) in A-element order; lane holds 32 of 64 j-values
    float av = ea[e];
    float eh0[16], eh1[16];
    #pragma unroll
    for (int t = 0; t < 16; ++t) {
        int K = t + ((t < 8) ? 0 : 8) + (hi ? 8 : 0);   // K in [0,32)
        eh0[t] = fmaxf(av * W1[K]      + b1[K],      0.0f);  // j = K
        eh1[t] = fmaxf(av * W1[32 + K] + b1[32 + K], 0.0f);  // j = 32+K
    }

    // gather source-node features (src < 8192 -> first s-slice of x, as in reference)
    int srcn = edge[e];
    float xr[16];
    {
        const float4* xp = (const float4*)(x + (size_t)srcn * 16);
        float4 a0 = xp[0], a1 = xp[1], a2 = xp[2], a3 = xp[3];
        xr[0]=a0.x; xr[1]=a0.y; xr[2]=a0.z; xr[3]=a0.w;
        xr[4]=a1.x; xr[5]=a1.y; xr[6]=a1.z; xr[7]=a1.w;
        xr[8]=a2.x; xr[9]=a2.y; xr[10]=a2.z; xr[11]=a2.w;
        xr[12]=a3.x; xr[13]=a3.y; xr[14]=a3.z; xr[15]=a3.w;
    }

    v8f c0 = {}, c1 = {}, c2 = {}, c3 = {};
    const v16bf* Bv = (const v16bf*)Bl;   // element i = 16 bf16; tile = 32 elements

    #pragma unroll
    for (int f = 0; f < 16; ++f) {
        #pragma unroll
        for (int jh = 0; jh < 2; ++jh) {
            v16bf A;
            #pragma unroll
            for (int t = 0; t < 16; ++t)
                A[t] = (__bf16)(xr[f] * (jh ? eh1[t] : eh0[t]));
            int tb = (f * 2 + jh) * 4;
            c0 = __builtin_amdgcn_wmma_f32_16x16x32_bf16(false, A, false, Bv[(tb+0)*32 + lane], (short)0, c0, false, false);
            c1 = __builtin_amdgcn_wmma_f32_16x16x32_bf16(false, A, false, Bv[(tb+1)*32 + lane], (short)0, c1, false, false);
            c2 = __builtin_amdgcn_wmma_f32_16x16x32_bf16(false, A, false, Bv[(tb+2)*32 + lane], (short)0, c2, false, false);
            c3 = __builtin_amdgcn_wmma_f32_16x16x32_bf16(false, A, false, Bv[(tb+3)*32 + lane], (short)0, c3, false, false);
        }
    }
    // bias slice: A = raw x row (K<16), zero otherwise
    {
        v16bf A;
        #pragma unroll
        for (int t = 0; t < 16; ++t) {
            float v = (t < 8) ? (hi ? xr[t + 8] : xr[t]) : 0.0f;
            A[t] = (__bf16)v;
        }
        c0 = __builtin_amdgcn_wmma_f32_16x16x32_bf16(false, A, false, Bv[(128+0)*32 + lane], (short)0, c0, false, false);
        c1 = __builtin_amdgcn_wmma_f32_16x16x32_bf16(false, A, false, Bv[(128+1)*32 + lane], (short)0, c1, false, false);
        c2 = __builtin_amdgcn_wmma_f32_16x16x32_bf16(false, A, false, Bv[(128+2)*32 + lane], (short)0, c2, false, false);
        c3 = __builtin_amdgcn_wmma_f32_16x16x32_bf16(false, A, false, Bv[(128+3)*32 + lane], (short)0, c3, false, false);
    }

    // scatter-add: C layout lane L -> N = L&15, VGPR r -> M = r + (hi?8:0)
    #pragma unroll
    for (int r = 0; r < 8; ++r) {
        int m = r + (hi ? 8 : 0);
        int d = edge[EE + base + m];                 // dst node
        float* ap = agg + (size_t)d * 64 + (lane & 15);
        atomicAdd(ap + 0,  c0[r]);
        atomicAdd(ap + 16, c1[r]);
        atomicAdd(ap + 32, c2[r]);
        atomicAdd(ap + 48, c3[r]);
    }
}

// ---------------- root transform + ReLU ----------------
__global__ __launch_bounds__(256) void conv_root_kernel(
    const float* __restrict__ x, const float* __restrict__ rootW,
    const float* __restrict__ convb, const float* __restrict__ agg,
    float* __restrict__ outc)
{
    int idx = blockIdx.x * 256 + threadIdx.x;       // over SNN*HD
    if (idx >= SNN * HD) return;
    int row = idx >> 6, h = idx & 63;
    const float* xr = x + (size_t)row * 16;
    float acc = agg[idx] + convb[h];
    #pragma unroll
    for (int f = 0; f < 16; ++f) acc += xr[f] * rootW[f * 64 + h];
    outc[idx] = fmaxf(acc, 0.0f);
}

// ---------------- GRU (4 sequential steps, batch = 8192 nodes) ----------------
__global__ __launch_bounds__(256) void gru_kernel(
    const float* __restrict__ xin, const float* __restrict__ h0,
    const float* __restrict__ Wih, const float* __restrict__ Whh,
    const float* __restrict__ bih, const float* __restrict__ bhh,
    float* __restrict__ ys)
{
    __shared__ float sWih[192 * 64];
    __shared__ float sWhh[192 * 64];
    __shared__ float sbih[192], sbhh[192];
    __shared__ float hbuf[64 * 64];
    __shared__ float xbuf[64 * 64];
    int tid = threadIdx.x;
    for (int i = tid; i < 192 * 64; i += 256) { sWih[i] = Wih[i]; sWhh[i] = Whh[i]; }
    for (int i = tid; i < 192; i += 256)      { sbih[i] = bih[i]; sbhh[i] = bhh[i]; }

    int nl = tid >> 2;                 // local node 0..63
    int sub = tid & 3;                 // 4 threads per node
    int hbase = sub * 16;
    int ng = blockIdx.x * 64 + nl;     // global node
    for (int i = 0; i < 16; ++i)
        hbuf[nl * 64 + hbase + i] = h0[(size_t)ng * 64 + hbase + i];
    __syncthreads();

    for (int s = 0; s < SS; ++s) {
        for (int i = 0; i < 16; ++i)
            xbuf[nl * 64 + hbase + i] = xin[((size_t)s * NN + ng) * 64 + hbase + i];
        __syncthreads();
        float hn[16];
        #pragma unroll
        for (int ii = 0; ii < 16; ++ii) {
            int hc = hbase + ii;
            float gir = sbih[hc], giz = sbih[64 + hc], gin = sbih[128 + hc];
            float ghr = sbhh[hc], ghz = sbhh[64 + hc], ghn = sbhh[128 + hc];
            const float* wr = &sWih[hc * 64];
            const float* wz = &sWih[(64 + hc) * 64];
            const float* wn = &sWih[(128 + hc) * 64];
            const float* vr = &sWhh[hc * 64];
            const float* vz = &sWhh[(64 + hc) * 64];
            const float* vn = &sWhh[(128 + hc) * 64];
            const float* xv = &xbuf[nl * 64];
            const float* hv = &hbuf[nl * 64];
            #pragma unroll 8
            for (int k = 0; k < 64; ++k) {
                float xk = xv[k], hk = hv[k];
                gir += xk * wr[k]; giz += xk * wz[k]; gin += xk * wn[k];
                ghr += hk * vr[k]; ghz += hk * vz[k]; ghn += hk * vn[k];
            }
            float r  = 1.0f / (1.0f + __expf(-(gir + ghr)));
            float z  = 1.0f / (1.0f + __expf(-(giz + ghz)));
            float t2 = tanhf(gin + r * ghn);
            hn[ii] = (1.0f - z) * t2 + z * hv[hc];
        }
        __syncthreads();
        #pragma unroll
        for (int ii = 0; ii < 16; ++ii) {
            hbuf[nl * 64 + hbase + ii] = hn[ii];
            ys[((size_t)s * NN + ng) * 64 + hbase + ii] = hn[ii];
        }
        __syncthreads();
    }
}

// ---------------- dueling-head GEMV (memory-bound: stream W once) ----------------
__global__ __launch_bounds__(256) void head_gemv(
    const float* __restrict__ W, const float* __restrict__ xflat,
    float* __restrict__ acc, int ncols)
{
    const int KS = FEATF / 16;                       // 32768 per k-slice
    int c = blockIdx.x, sl = blockIdx.y;
    const float* w  = W + (size_t)c * FEATF + (size_t)sl * KS;
    const float* x0 = xflat + (size_t)sl * KS;
    float a0 = 0.f, a1 = 0.f, a2 = 0.f, a3 = 0.f;
    for (int k = threadIdx.x; k < KS; k += 256) {
        float wv = w[k];
        a0 += wv * x0[k];
        a1 += wv * x0[FEATF + k];
        a2 += wv * x0[2 * FEATF + k];
        a3 += wv * x0[3 * FEATF + k];
    }
    __shared__ float red[256 * 4];
    int t = threadIdx.x;
    red[t] = a0; red[256 + t] = a1; red[512 + t] = a2; red[768 + t] = a3;
    __syncthreads();
    for (int off = 128; off > 0; off >>= 1) {
        if (t < off) {
            red[t] += red[t + off];           red[256 + t] += red[256 + t + off];
            red[512 + t] += red[512 + t + off]; red[768 + t] += red[768 + t + off];
        }
        __syncthreads();
    }
    if (t == 0) {
        atomicAdd(&acc[0 * ncols + c], red[0]);
        atomicAdd(&acc[1 * ncols + c], red[256]);
        atomicAdd(&acc[2 * ncols + c], red[512]);
        atomicAdd(&acc[3 * ncols + c], red[768]);
    }
}

// ---------------- value MLP + dueling combine (48 outputs) ----------------
__global__ __launch_bounds__(64) void finalize_kernel(
    const float* __restrict__ advacc, const float* __restrict__ v1acc,
    const float* __restrict__ adv_b, const float* __restrict__ v1b,
    const float* __restrict__ v2W, const float* __restrict__ v2b,
    const float* __restrict__ v3W, const float* __restrict__ v3b,
    float* __restrict__ out)
{
    __shared__ float v1[4][64], v2[4][64], vv[4];
    int t = threadIdx.x;
    for (int s = 0; s < 4; ++s) v1[s][t] = fmaxf(v1acc[s * 64 + t] + v1b[t], 0.0f);
    __syncthreads();
    for (int s = 0; s < 4; ++s) {
        float a = v2b[t];
        #pragma unroll 8
        for (int k = 0; k < 64; ++k) a += v1[s][k] * v2W[t * 64 + k];
        v2[s][t] = fmaxf(a, 0.0f);
    }
    __syncthreads();
    if (t < 4) {
        float a = v3b[0];
        for (int k = 0; k < 64; ++k) a += v2[t][k] * v3W[k];
        vv[t] = a;
    }
    __syncthreads();
    if (t < 48) {
        int s = t / 12, w = t % 12, g = w / 3, j = w % 3;
        float a0 = fmaxf(advacc[s * 12 + g * 3 + 0] + adv_b[g * 3 + 0], 0.0f);
        float a1 = fmaxf(advacc[s * 12 + g * 3 + 1] + adv_b[g * 3 + 1], 0.0f);
        float a2 = fmaxf(advacc[s * 12 + g * 3 + 2] + adv_b[g * 3 + 2], 0.0f);
        float aj = (j == 0) ? a0 : ((j == 1) ? a1 : a2);
        float mean = (a0 + a1 + a2) * (1.0f / 3.0f);
        out[t] = vv[s] + aj - mean;
    }
}

extern "C" void kernel_launch(void* const* d_in, const int* in_sizes, int n_in,
                              void* d_out, int out_size, void* d_ws, size_t ws_size,
                              hipStream_t stream)
{
    const float* x     = (const float*)d_in[0];
    const int*   edge  = (const int*)d_in[1];
    const float* ea    = (const float*)d_in[2];
    const float* h0    = (const float*)d_in[3];
    const float* W1    = (const float*)d_in[4];
    const float* b1    = (const float*)d_in[5];
    const float* W2    = (const float*)d_in[6];
    const float* b2    = (const float*)d_in[7];
    const float* rootW = (const float*)d_in[8];
    const float* convb = (const float*)d_in[9];
    const float* Wih   = (const float*)d_in[10];
    const float* Whh   = (const float*)d_in[11];
    const float* bih   = (const float*)d_in[12];
    const float* bhh   = (const float*)d_in[13];
    const float* advW  = (const float*)d_in[14];
    const float* advb  = (const float*)d_in[15];
    const float* v1W   = (const float*)d_in[16];
    const float* v1b   = (const float*)d_in[17];
    const float* v2W   = (const float*)d_in[18];
    const float* v2b   = (const float*)d_in[19];
    const float* v3W   = (const float*)d_in[20];
    const float* v3b   = (const float*)d_in[21];
    float* out = (float*)d_out;

    char* ws = (char*)d_ws;
    float*  agg    = (float*)(ws);                    //  8 MB  (32768*64 f32)
    float*  convo  = (float*)(ws + 8388608);          //  8 MB
    float*  ys     = (float*)(ws + 16777216);         //  8 MB
    __bf16* Bbuf   = (__bf16*)(ws + 25165824);        //  132 KB packed B
    float*  advacc = (float*)(ws + 25303040);         //  48 + 256 floats
    float*  v1acc  = advacc + 48;

    zero_kernel<<<2048, 256, 0, stream>>>(agg, SNN * HD);
    zero_kernel<<<1, 256, 0, stream>>>(advacc, 48 + 256);
    prep_b_kernel<<<(BELEM + 255) / 256, 256, 0, stream>>>(W2, b2, Bbuf);
    nnconv_kernel<<<EE / 128, 256, 0, stream>>>(x, edge, ea, W1, b1, Bbuf, agg);
    conv_root_kernel<<<(SNN * HD) / 256, 256, 0, stream>>>(x, rootW, convb, agg, convo);
    gru_kernel<<<NN / 64, 256, 0, stream>>>(convo, h0, Wih, Whh, bih, bhh, ys);
    head_gemv<<<dim3(12, 16), 256, 0, stream>>>(advW, ys, advacc, 12);
    head_gemv<<<dim3(64, 16), 256, 0, stream>>>(v1W, ys, v1acc, 64);
    finalize_kernel<<<1, 64, 0, stream>>>(advacc, v1acc, advb, v1b, v2W, v2b, v3W, v3b, out);
}
